// SNNComponent_53979148976315
// MI455X (gfx1250) — matrix operand order repaired and checked by reading, not compiled, not run on hardware
//
#include <hip/hip_runtime.h>
#include <cstdint>

typedef __attribute__((ext_vector_type(16))) _Float16 v16h;
typedef __attribute__((ext_vector_type(8)))  _Float16 v8h;
typedef __attribute__((ext_vector_type(8)))  float    v8f;

static constexpr float kThresh = 1.0f;
static constexpr float kBeta   = 0.95f;

#define NT 4                        // 16-col n-tiles per wave
#define WAVES 4
#define BLOCK (WAVES * 32)          // 128 threads
#define TILE_COLS (WAVES * NT * 16) // 256 columns per block

// ---------------------------------------------------------------------------
// Setup / elementwise kernels
// ---------------------------------------------------------------------------
__global__ void cvt_f32_to_f16(const float* __restrict__ src, _Float16* __restrict__ dst, int n) {
  int i = blockIdx.x * blockDim.x + threadIdx.x;
  if (i < n) dst[i] = (_Float16)src[i];
}

__global__ void zero_f32(float* __restrict__ p, int n) {
  int i = blockIdx.x * blockDim.x + threadIdx.x;
  if (i < n) p[i] = 0.0f;
}

// Layer-1 LIF: mem1 = beta*mem1 + cur1 - (mem1_prev > thr); spk1 = (mem1 > thr)
__global__ void lif1_step(const float* __restrict__ cur1, float* __restrict__ mem1,
                          _Float16* __restrict__ spk1, int n) {
  int i = blockIdx.x * blockDim.x + threadIdx.x;
  if (i >= n) return;
  float mo    = mem1[i];
  float reset = (mo > kThresh) ? 1.0f : 0.0f;
  float mn    = kBeta * mo + cur1[i] - reset * kThresh;
  mem1[i] = mn;
  spk1[i] = (mn > kThresh) ? (_Float16)1.0f : (_Float16)0.0f;
}

// ---------------------------------------------------------------------------
// Fragment helpers
// ---------------------------------------------------------------------------
__device__ __forceinline__ v16h combine16(v8h lo, v8h hi) {
  v16h r;
#pragma unroll
  for (int i = 0; i < 8; ++i) { r[i] = lo[i]; r[i + 8] = hi[i]; }
  return r;
}

// 16-bit B fragment 32x16 (KxN), weights stored [N,K] row-major:
// lanes 0-15 col n=lane with K=k+[0..7]+[16..23]; lanes 16-31 col n=lane-16, K=k+[8..15]+[24..31].
__device__ __forceinline__ v16h load_b32x16(const _Float16* __restrict__ Wrow, int ldk,
                                            int k, int lane) {
  const _Float16* p = Wrow + (size_t)(lane & 15) * ldk + k + ((lane & 16) ? 8 : 0);
  return combine16(*(const v8h*)p, *(const v8h*)(p + 16));
}

// A fragment from LDS (16-bit A 16x32 layout):
// lanes 0-15: row=lane, halfs [0..7]+[16..23]; lanes 16-31: row=lane-16, [8..15]+[24..31].
__device__ __forceinline__ v16h load_a_lds(const _Float16* sm, int aoff) {
  const _Float16* ap = sm + aoff;
  return combine16(*(const v8h*)ap, *(const v8h*)(ap + 16));
}

// Minimal cross-wave sync for the LDS tile pipeline:
//  - asynccnt 0: my slice of the incoming tile has landed in LDS
//  - dscnt 0:    my reads of the outgoing buffer are done (already 0: WMMAs consumed them)
//  - split workgroup barrier: everyone else's too.
// Deliberately does NOT drain loadcnt (B prefetches keep flying across the barrier).
__device__ __forceinline__ void sync_tile() {
  asm volatile(
      "s_wait_asynccnt 0x0\n\t"
      "s_wait_dscnt 0x0\n\t"
      "s_barrier_signal -1\n\t"
      "s_barrier_wait -1" ::: "memory");
}

// ---------------------------------------------------------------------------
// GEMM with LDS-staged A (async copy, ping-pong buffered) + optional LIF epilogue.
//   out = A[M,K] @ W[N,K]^T + bias                   (LIF=false)
//   mem = beta*mem + ^ - (mem>thr); out=(mem>thr)    (LIF=true)
// One block: 16 rows x 256 cols (4 waves). One wave: 16x64 (NT=4 accumulators).
// Requires K % 64 == 0 and K >= 64.
// ---------------------------------------------------------------------------
template <bool LIF>
__global__ __launch_bounds__(BLOCK)
void gemm_lds_wmma(const _Float16* __restrict__ A, const _Float16* __restrict__ W,
                   const float* __restrict__ bias, float* __restrict__ mem,
                   float* __restrict__ out, int K, int N, int nblocks) {
  __shared__ __align__(16) _Float16 smem[2][16 * 32];  // 2 x 1KB A tile buffers

  const int tid  = (int)threadIdx.x;
  const int lane = tid & 31;
  const int wave = tid >> 5;
  const int mt   = (int)blockIdx.x / nblocks;   // block-uniform
  const int nblk = (int)blockIdx.x % nblocks;
  const int nt0  = (nblk * WAVES + wave) * NT;  // wave-uniform

  const _Float16* Ab = A + (size_t)mt * 16 * K;

  // Async-copy lane assignment: 128 lanes x 8B cover the 16x32-f16 (1KB) tile.
  const int  crow  = tid >> 3;        // 0..15 row
  const int  cbyte = (tid & 7) * 8;   // 8B chunk within 64B row
  const char* gsrc = (const char*)(Ab + (size_t)crow * K) + cbyte;
  const uint32_t lbase =
      (uint32_t)(uintptr_t)(&smem[0][0]) + (uint32_t)(crow * 64 + cbyte);

  auto issue_copy = [&](int buf, int k) {
    const char* g = gsrc + (size_t)k * sizeof(_Float16);
    uint32_t    l = lbase + (uint32_t)(buf * 16 * 32 * (int)sizeof(_Float16));
    asm volatile("global_load_async_to_lds_b64 %0, %1, off"
                 :: "v"(l), "v"(g) : "memory");
  };

  const _Float16* Bw[NT];
#pragma unroll
  for (int j = 0; j < NT; ++j) Bw[j] = W + (size_t)(nt0 + j) * 16 * K;

  v8f acc[NT];
  v8f z = {};
#pragma unroll
  for (int j = 0; j < NT; ++j) acc[j] = z;

  const int aoff = (lane & 15) * 32 + ((lane & 16) ? 8 : 0);

  // Prologue: start A(k=0) async copy, preload B(k=0).
  issue_copy(0, 0);
  v16h b0[NT], b1[NT];
#pragma unroll
  for (int j = 0; j < NT; ++j) b0[j] = load_b32x16(Bw[j], K, 0, lane);

  // Steady state: two 32-wide phases per iteration, ping-pong regs + LDS bufs,
  // no conditionals, no register rotation.
  int k = 0;
  for (; k + 64 < K; k += 64) {
    // phase 0: consume smem[0]/b0; prefetch smem[1]/b1 for k+32
    sync_tile();
    v16h a = load_a_lds(&smem[0][0], aoff);
    issue_copy(1, k + 32);
#pragma unroll
    for (int j = 0; j < NT; ++j) b1[j] = load_b32x16(Bw[j], K, k + 32, lane);
#pragma unroll
    for (int j = 0; j < NT; ++j)
      acc[j] = __builtin_amdgcn_wmma_f32_16x16x32_f16(false, a, false, b0[j],
                                                      (short)0, acc[j], false, false);

    // phase 1: consume smem[1]/b1; prefetch smem[0]/b0 for k+64
    sync_tile();
    a = load_a_lds(&smem[1][0], aoff);
    issue_copy(0, k + 64);
#pragma unroll
    for (int j = 0; j < NT; ++j) b0[j] = load_b32x16(Bw[j], K, k + 64, lane);
#pragma unroll
    for (int j = 0; j < NT; ++j)
      acc[j] = __builtin_amdgcn_wmma_f32_16x16x32_f16(false, a, false, b1[j],
                                                      (short)0, acc[j], false, false);
  }

  // Tail: k == K-64; phase 0 prefetches the final slice, phase 1 issues nothing.
  sync_tile();
  v16h a = load_a_lds(&smem[0][0], aoff);
  issue_copy(1, k + 32);
#pragma unroll
  for (int j = 0; j < NT; ++j) b1[j] = load_b32x16(Bw[j], K, k + 32, lane);
#pragma unroll
  for (int j = 0; j < NT; ++j)
    acc[j] = __builtin_amdgcn_wmma_f32_16x16x32_f16(false, a, false, b0[j],
                                                    (short)0, acc[j], false, false);
  sync_tile();
  a = load_a_lds(&smem[1][0], aoff);
#pragma unroll
  for (int j = 0; j < NT; ++j)
    acc[j] = __builtin_amdgcn_wmma_f32_16x16x32_f16(false, a, false, b1[j],
                                                    (short)0, acc[j], false, false);

  // Epilogue. D layout: VGPR r -> row r + (lane>=16 ? 8 : 0); lane&15 -> col.
  const int mbase = mt * 16 + ((lane & 16) ? 8 : 0);
#pragma unroll
  for (int j = 0; j < NT; ++j) {
    const int   ncol = (nt0 + j) * 16 + (lane & 15);
    const float bj   = bias[ncol];
#pragma unroll
    for (int r = 0; r < 8; ++r) {
      const size_t idx = (size_t)(mbase + r) * N + ncol;
      float v = acc[j][r] + bj;
      if constexpr (LIF) {
        float mo    = mem[idx];
        float reset = (mo > kThresh) ? 1.0f : 0.0f;
        float mn    = kBeta * mo + v - reset * kThresh;
        mem[idx] = mn;
        out[idx] = (mn > kThresh) ? 1.0f : 0.0f;
      } else {
        out[idx] = v;
      }
    }
  }
}

// ---------------------------------------------------------------------------
// Host launcher
// ---------------------------------------------------------------------------
extern "C" void kernel_launch(void* const* d_in, const int* in_sizes, int n_in,
                              void* d_out, int out_size, void* d_ws, size_t ws_size,
                              hipStream_t stream) {
  (void)n_in; (void)ws_size;
  const float* x  = (const float*)d_in[0];
  const float* W1 = (const float*)d_in[1];
  const float* b1 = (const float*)d_in[2];
  const float* W2 = (const float*)d_in[3];
  const float* b2 = (const float*)d_in[4];

  const int H   = in_sizes[2];            // 4096
  const int OUT = in_sizes[4];            // 1024
  const int IN  = in_sizes[1] / H;        // 2048
  const int B   = in_sizes[0] / IN;       // 512
  const int T   = out_size / (B * OUT);   // 100

  // --- carve workspace (256B aligned) ---
  uintptr_t p = (uintptr_t)d_ws;
  auto carve = [&](size_t bytes) -> void* {
    uintptr_t q = (p + 255) & ~(uintptr_t)255;
    p = q + bytes;
    return (void*)q;
  };
  _Float16* xh   = (_Float16*)carve((size_t)B * IN * sizeof(_Float16));
  _Float16* W1h  = (_Float16*)carve((size_t)H * IN * sizeof(_Float16));
  _Float16* W2h  = (_Float16*)carve((size_t)OUT * H * sizeof(_Float16));
  _Float16* spk1 = (_Float16*)carve((size_t)B * H * sizeof(_Float16));
  float*    cur1 = (float*)   carve((size_t)B * H * sizeof(float));
  float*    mem1 = (float*)   carve((size_t)B * H * sizeof(float));
  float*    mem2 = (float*)   carve((size_t)B * OUT * sizeof(float));

  auto nblk = [](int n) { return (n + 255) / 256; };

  // Per-launch setup: f16 conversion + state zeroing (deterministic across replays).
  cvt_f32_to_f16<<<nblk(B * IN), 256, 0, stream>>>(x,  xh,  B * IN);
  cvt_f32_to_f16<<<nblk(H * IN), 256, 0, stream>>>(W1, W1h, H * IN);
  cvt_f32_to_f16<<<nblk(OUT * H), 256, 0, stream>>>(W2, W2h, OUT * H);
  zero_f32<<<nblk(B * H),   256, 0, stream>>>(mem1, B * H);
  zero_f32<<<nblk(B * OUT), 256, 0, stream>>>(mem2, B * OUT);

  // Loop-invariant GEMM1: cur1 = x @ W1^T + b1  (512 blocks)
  const int nb1 = H / TILE_COLS;
  gemm_lds_wmma<false><<<(B / 16) * nb1, BLOCK, 0, stream>>>(
      xh, W1h, b1, nullptr, cur1, IN, H, nb1);

  // Time loop: LIF1 -> (GEMM2 + fused LIF2 -> d_out[t])  (128 blocks/step)
  const int nb2 = OUT / TILE_COLS;
  float* out = (float*)d_out;
  for (int t = 0; t < T; ++t) {
    lif1_step<<<nblk(B * H), 256, 0, stream>>>(cur1, mem1, spk1, B * H);
    gemm_lds_wmma<true><<<(B / 16) * nb2, BLOCK, 0, stream>>>(
        spk1, W2h, b2, mem2, out + (size_t)t * B * OUT, H, OUT, nb2);
  }
}
